// Drug_3d_Encoder_9826885173484
// MI455X (gfx1250) — compile-verified, matrix operand order checked by reading, not run on hardware
//
#include <hip/hip_runtime.h>
#include <hip/hip_bf16.h>

// ---------------------------------------------------------------------------
// Drug3D GNN encoder for gfx1250 (MI455X).
// Dense matmuls run through v_wmma_f32_16x16x32_bf16 (fp32 accumulate).
// Global->LDS staging uses CDNA5 async loads (global_load_async_to_lds_b128,
// ASYNCcnt) with double-buffered LDS tiles so DMA overlaps the WMMA stream.
// Gathers/scatters use global_atomic_add_f32; GraphNorm uses a fused
// sum/sumsq column reduction.
// ---------------------------------------------------------------------------

typedef __attribute__((ext_vector_type(16))) __bf16 v16bf;
typedef __attribute__((ext_vector_type(8)))  __bf16 v8bf;
typedef __attribute__((ext_vector_type(4)))  __bf16 v4bf;
typedef __attribute__((ext_vector_type(8)))  float  v8f;

#define DD   256
#define DD2  512
#define NNODE 65536
#define NEDGE 131072
#define NANG  262144
#define NGRAPH 1024
#define NFC   1024

// ------------------------------ GEMM (WMMA) --------------------------------
// C[M,N] = A[M,K] (bf16, row-major) @ Bt[N,K]^T (bf16, N-major) + bias, relu?
// block = 256 threads (8 waves), tile 128x128, K-step 32, double-buffered LDS.
#define BM 128
#define BN 128
#define BK 32
#define LDT 40   // padded LDS row stride in halves (80B; keeps 16B alignment)

__device__ __forceinline__ v16bf frag_ld(const __bf16* p) {
  v8bf lo = *(const v8bf*)p;         // K = kb .. kb+7
  v8bf hi = *(const v8bf*)(p + 16);  // K = kb+16 .. kb+23
  return __builtin_shufflevector(lo, hi, 0,1,2,3,4,5,6,7,8,9,10,11,12,13,14,15);
}

// CDNA5 async DMA: 16B global -> LDS, tracked with ASYNCcnt (no VGPR staging).
// Low 32 bits of a generic shared pointer are the LDS byte address (the
// shared aperture lives in the high bits).
__device__ __forceinline__ void async_b128(const __bf16* gp, __bf16* lp) {
  unsigned lofs = (unsigned)(uintptr_t)lp;
  unsigned long long ga = (unsigned long long)(uintptr_t)gp;
  asm volatile("global_load_async_to_lds_b128 %0, %1, off"
               :: "v"(lofs), "v"(ga) : "memory");
}

__global__ __launch_bounds__(256)
void k_gemm_bf16(const __bf16* __restrict__ A, const __bf16* __restrict__ Bt,
                 const float* __restrict__ bias, float* __restrict__ C,
                 int M, int N, int K, int relu)
{
  __shared__ __attribute__((aligned(16))) __bf16 sA[2][BM * LDT];
  __shared__ __attribute__((aligned(16))) __bf16 sB[2][BN * LDT];

  const int tid  = threadIdx.x;
  const int lane = tid & 31;
  const int wave = tid >> 5;      // 0..7
  const int wr   = wave >> 2;     // 0..1 -> 64-row half
  const int wc   = wave & 3;      // 0..3 -> 32-col quarter
  const int rowBase = blockIdx.x * BM;
  const int colBase = blockIdx.y * BN;

  // per-thread staging assignment: 16B chunk (tid) covers rows 0..63,
  // chunk (tid+256) covers rows 64..127; 4 chunks of 8 halves per 32-half row.
  const int r0 = tid >> 2, ch = tid & 3;
  const int r1 = r0 + 64;

  v8f acc[4][2];
#pragma unroll
  for (int mi = 0; mi < 4; ++mi)
#pragma unroll
    for (int ni = 0; ni < 2; ++ni)
#pragma unroll
      for (int j = 0; j < 8; ++j) acc[mi][ni][j] = 0.f;

  const int mrow = lane & 15;
  const int kb   = (lane < 16) ? 0 : 8;   // ISA 16-bit A/B fragment layout

  // prologue: fill buffer 0
  {
    const __bf16* Ab = A  + (size_t)rowBase * K + ch * 8;
    const __bf16* Bb = Bt + (size_t)colBase * K + ch * 8;
    async_b128(Ab + (size_t)r0 * K, &sA[0][r0 * LDT + ch * 8]);
    async_b128(Ab + (size_t)r1 * K, &sA[0][r1 * LDT + ch * 8]);
    async_b128(Bb + (size_t)r0 * K, &sB[0][r0 * LDT + ch * 8]);
    async_b128(Bb + (size_t)r1 * K, &sB[0][r1 * LDT + ch * 8]);
  }

  int buf = 0;
  for (int k0 = 0; k0 < K; k0 += BK) {
    const bool has_next = (k0 + BK) < K;
    if (has_next) {  // overlap next slice DMA with this slice's WMMAs
      const int nb = buf ^ 1;
      const __bf16* Ab = A  + (size_t)rowBase * K + k0 + BK + ch * 8;
      const __bf16* Bb = Bt + (size_t)colBase * K + k0 + BK + ch * 8;
      async_b128(Ab + (size_t)r0 * K, &sA[nb][r0 * LDT + ch * 8]);
      async_b128(Ab + (size_t)r1 * K, &sA[nb][r1 * LDT + ch * 8]);
      async_b128(Bb + (size_t)r0 * K, &sB[nb][r0 * LDT + ch * 8]);
      async_b128(Bb + (size_t)r1 * K, &sB[nb][r1 * LDT + ch * 8]);
      // async ops complete in order: <=4 outstanding means current buf ready
      asm volatile("s_wait_asynccnt 0x4" ::: "memory");
    } else {
      asm volatile("s_wait_asynccnt 0x0" ::: "memory");
    }
    __syncthreads();

    v16bf af[4], bq[2];
#pragma unroll
    for (int mi = 0; mi < 4; ++mi)
      af[mi] = frag_ld(&sA[buf][(wr * 64 + mi * 16 + mrow) * LDT + kb]);
#pragma unroll
    for (int ni = 0; ni < 2; ++ni)
      bq[ni] = frag_ld(&sB[buf][(wc * 32 + ni * 16 + mrow) * LDT + kb]);

#pragma unroll
    for (int mi = 0; mi < 4; ++mi)
#pragma unroll
      for (int ni = 0; ni < 2; ++ni)
        acc[mi][ni] = __builtin_amdgcn_wmma_f32_16x16x32_bf16(
            false, af[mi], false, bq[ni], (short)0, acc[mi][ni], false, false);

    __syncthreads();  // all waves done reading buf before it is refilled
    buf ^= 1;
  }

  // epilogue: VGPR r holds M=r (lanes 0-15) / M=r+8 (lanes 16-31), N=lane&15
  const int nlane = lane & 15;
  const int mofs  = (lane < 16) ? 0 : 8;
#pragma unroll
  for (int mi = 0; mi < 4; ++mi)
#pragma unroll
    for (int ni = 0; ni < 2; ++ni) {
      const int n  = colBase + wc * 32 + ni * 16 + nlane;
      const int mb = rowBase + wr * 64 + mi * 16 + mofs;
      const float bv = bias ? bias[n] : 0.f;
#pragma unroll
      for (int r = 0; r < 8; ++r) {
        float v = acc[mi][ni][r] + bv;
        if (relu) v = fmaxf(v, 0.f);
        C[(size_t)(mb + r) * N + n] = v;
      }
    }
}

// --------------------------- support kernels -------------------------------

__global__ __launch_bounds__(256)
void k_cvt_bf16(const float* __restrict__ in, __bf16* __restrict__ out, long n4) {
  long i = (long)blockIdx.x * blockDim.x + threadIdx.x;
  long stride = (long)gridDim.x * blockDim.x;
  for (; i < n4; i += stride) {
    float4 v = ((const float4*)in)[i];
    v4bf o;
    o[0] = (__bf16)v.x; o[1] = (__bf16)v.y; o[2] = (__bf16)v.z; o[3] = (__bf16)v.w;
    ((v4bf*)out)[i] = o;
  }
}

// wt[n*K + k] = (bf16) w[k*N + n]   (weights are tiny; coalesced on output)
__global__ __launch_bounds__(256)
void k_cvtT(const float* __restrict__ w, __bf16* __restrict__ wt, int K, int N) {
  int idx = blockIdx.x * 256 + threadIdx.x;
  if (idx >= K * N) return;
  int n = idx / K, k = idx - n * K;
  wt[idx] = (__bf16)w[(size_t)k * N + n];
}

__global__ __launch_bounds__(256)
void k_zero(float* __restrict__ p, long n) {
  long i = (long)blockIdx.x * blockDim.x + threadIdx.x;
  long stride = (long)gridDim.x * blockDim.x;
  for (; i < n; i += stride) p[i] = 0.f;
}

// out[row, :] = sum_f emb[f, idx[row,f], :]
__global__ __launch_bounds__(256)
void k_embed(const float* __restrict__ emb, const int* __restrict__ idx,
             float* __restrict__ out, int F, int V) {
  const int row = blockIdx.x, d = threadIdx.x;
  float s = 0.f;
  for (int f = 0; f < F; ++f) {
    int v = idx[row * F + f];
    s += emb[((size_t)f * V + v) * DD + d];
  }
  out[(size_t)row * DD + d] = s;
}

// h[dst] += relu(xsrc[src] + lin[e])   (h pre-initialized to x)
__global__ __launch_bounds__(256)
void k_scatter(const float* __restrict__ xsrc, const float* __restrict__ lin,
               const int* __restrict__ ei, int nE, float* __restrict__ h) {
  const int e = blockIdx.x, d = threadIdx.x;
  const int s = ei[e], t = ei[nE + e];
  float v = fmaxf(xsrc[(size_t)s * DD + d] + lin[(size_t)e * DD + d], 0.f);
  atomicAdd(&h[(size_t)t * DD + d], v);
}

// ang1[a, :] = relu(attr[a] * w1 + b1)
__global__ __launch_bounds__(256)
void k_ang1(const float* __restrict__ attr, const float* __restrict__ w1,
            const float* __restrict__ b1, float* __restrict__ out) {
  const int a = blockIdx.x, d = threadIdx.x;
  out[(size_t)a * DD + d] = fmaxf(attr[a] * w1[d] + b1[d], 0.f);
}

__global__ __launch_bounds__(256)
void k_layernorm(float* __restrict__ x, const float* __restrict__ w,
                 const float* __restrict__ b) {
  const int row = blockIdx.x, d = threadIdx.x;
  __shared__ float red[8];
  float v = x[(size_t)row * DD + d];
  float s = v;
  for (int o = 16; o; o >>= 1) s += __shfl_down(s, o, 32);
  if ((d & 31) == 0) red[d >> 5] = s;
  __syncthreads();
  float m = 0.f;
  for (int i = 0; i < 8; ++i) m += red[i];
  m *= (1.f / 256.f);
  __syncthreads();
  float c = v - m, q = c * c;
  for (int o = 16; o; o >>= 1) q += __shfl_down(q, o, 32);
  if ((d & 31) == 0) red[d >> 5] = q;
  __syncthreads();
  float var = 0.f;
  for (int i = 0; i < 8; ++i) var += red[i];
  var *= (1.f / 256.f);
  x[(size_t)row * DD + d] = c * rsqrtf(var + 1e-5f) * w[d] + b[d];
}

// per-column sum & sumsq (256 rows per block, atomic merge)
__global__ __launch_bounds__(256)
void k_colstats(const float* __restrict__ x, float* __restrict__ stat, int M) {
  const int d = threadIdx.x;
  const int r0 = blockIdx.x * 256;
  float s = 0.f, q = 0.f;
  for (int r = r0; r < r0 + 256; ++r) {
    float v = x[(size_t)r * DD + d];
    s += v; q += v * v;
  }
  atomicAdd(&stat[d], s);
  atomicAdd(&stat[DD + d], q);
  (void)M;
}

// GraphNorm apply (+optional relu) then residual: res = res + y
__global__ __launch_bounds__(256)
void k_gn_apply(const float* __restrict__ xn, const float* __restrict__ stat,
                const float* __restrict__ w, const float* __restrict__ b,
                const float* __restrict__ ms, float* __restrict__ res,
                int M, int relu) {
  const int row = blockIdx.x, d = threadIdx.x;
  const float invM = 1.f / (float)M;
  const float m   = stat[d] * invM;
  const float ex2 = stat[DD + d] * invM;
  const float msd = ms[d];
  // E[(x - ms*m)^2] = E[x^2] - 2*ms*m^2 + ms^2*m^2
  const float var = ex2 - 2.f * msd * m * m + msd * msd * m * m;
  float c = xn[(size_t)row * DD + d] - msd * m;
  float y = w[d] * c * rsqrtf(var + 1e-5f) + b[d];
  if (relu) y = fmaxf(y, 0.f);
  res[(size_t)row * DD + d] += y;
}

__global__ __launch_bounds__(256)
void k_pool(const float* __restrict__ x, const int* __restrict__ batch,
            float* __restrict__ sums, float* __restrict__ cnt) {
  const int r = blockIdx.x, d = threadIdx.x;
  const int g = batch[r];
  atomicAdd(&sums[(size_t)g * DD + d], x[(size_t)r * DD + d]);
  if (d == 0) atomicAdd(&cnt[g], 1.f);
}

__global__ __launch_bounds__(256)
void k_pooldiv(float* __restrict__ sums, const float* __restrict__ cnt) {
  const int g = blockIdx.x, d = threadIdx.x;
  sums[(size_t)g * DD + d] /= fmaxf(cnt[g], 1.f);
}

// ------------------------------ host side ----------------------------------

static inline void gemm(const __bf16* A, const __bf16* Bt, const float* bias,
                        float* C, int M, int N, int K, int relu, hipStream_t s) {
  dim3 grid(M / BM, N / BN);
  k_gemm_bf16<<<grid, 256, 0, s>>>(A, Bt, bias, C, M, N, K, relu);
}
static inline void cvt(const float* in, __bf16* out, long n, hipStream_t s) {
  k_cvt_bf16<<<2048, 256, 0, s>>>(in, out, n / 4);
}
static inline void cvtT(const float* w, __bf16* wt, int K, int N, hipStream_t s) {
  k_cvtT<<<(K * N + 255) / 256, 256, 0, s>>>(w, wt, K, N);
}
static inline void zero(float* p, long n, hipStream_t s) {
  k_zero<<<256, 256, 0, s>>>(p, n);
}

extern "C" void kernel_launch(void* const* d_in, const int* in_sizes, int n_in,
                              void* d_out, int out_size, void* d_ws, size_t ws_size,
                              hipStream_t stream) {
  (void)in_sizes; (void)n_in; (void)out_size; (void)ws_size;
  // Input order: JAX pytree flatten of setup_inputs() dict (alphabetical keys;
  // 'params' nested dict also alphabetical).
  const int*   atom_edge_attr = (const int*)d_in[0];   // [E,3]
  const int*   atom_ei        = (const int*)d_in[1];   // [2,E]
  const int*   atom_x         = (const int*)d_in[2];   // [N,9]
  const int*   batch          = (const int*)d_in[3];   // [N]
  const float* bond_attr      = (const float*)d_in[4]; // [A,1]
  const int*   bond_ei        = (const int*)d_in[5];   // [2,A]
  const float* ang_b1   = (const float*)d_in[6];
  const float* ang_b2   = (const float*)d_in[7];
  const float* ang_w1   = (const float*)d_in[8];
  const float* ang_w2   = (const float*)d_in[9];
  const float* atom_emb = (const float*)d_in[10];
  const float* bemb0    = (const float*)d_in[11];
  const float* bembL    = (const float*)d_in[12];
  const float* fc1_b    = (const float*)d_in[13];
  const float* fc1_w    = (const float*)d_in[14];
  const float* fc2_b    = (const float*)d_in[15];
  const float* fc2_w    = (const float*)d_in[16];
  const float* ga_b1    = (const float*)d_in[17];
  const float* ga_b2    = (const float*)d_in[18];
  const float* ga_lin_b = (const float*)d_in[19];
  const float* ga_lin_w = (const float*)d_in[20];
  const float* ga_w1    = (const float*)d_in[21];
  const float* ga_w2    = (const float*)d_in[22];
  const float* gb_b1    = (const float*)d_in[23];
  const float* gb_b2    = (const float*)d_in[24];
  const float* gb_lin_b = (const float*)d_in[25];
  const float* gb_lin_w = (const float*)d_in[26];
  const float* gb_w1    = (const float*)d_in[27];
  const float* gb_w2    = (const float*)d_in[28];
  const float* gn_a_b   = (const float*)d_in[29];
  const float* gn_a_ms  = (const float*)d_in[30];
  const float* gn_a_w   = (const float*)d_in[31];
  const float* gn_b_b   = (const float*)d_in[32];
  const float* gn_b_ms  = (const float*)d_in[33];
  const float* gn_b_w   = (const float*)d_in[34];
  const float* ln_a_b   = (const float*)d_in[35];
  const float* ln_a_w   = (const float*)d_in[36];
  const float* ln_b_b   = (const float*)d_in[37];
  const float* ln_b_w   = (const float*)d_in[38];

  // ---- carve workspace ----
  char* base = (char*)d_ws;
  size_t off = 0;
  auto carve = [&](size_t bytes) -> char* {
    char* p = base + off;
    off += (bytes + 255) & ~(size_t)255;
    return p;
  };
  float* xbuf   = (float*)carve((size_t)NNODE * DD * 4);       // atom features
  float* ehbuf  = (float*)carve((size_t)NEDGE * DD * 4);       // bond features
  float* ceabuf = (float*)carve((size_t)NEDGE * DD * 4);       // per-layer bond embed
  float* tA     = (float*)carve((size_t)NANG  * DD * 4);       // big temp 1
  float* tB     = (float*)carve((size_t)NANG  * DD * 4);       // big temp 2
  __bf16* abuf  = (__bf16*)carve((size_t)NANG * DD * 2);       // bf16 activations
  __bf16* wbuf  = (__bf16*)carve((size_t)1024 * 1024 * 2);     // bf16 weights (T)
  float* gstat  = (float*)carve(2 * DD * 4);                   // colsum / colsumsq
  float* sums   = (float*)carve((size_t)NGRAPH * DD * 4);      // pooled sums
  float* cnt    = (float*)carve((size_t)NGRAPH * 4);           // pooled counts

  // ---- embeddings ----
  k_embed<<<NNODE, 256, 0, stream>>>(atom_emb, atom_x, xbuf, 9, 119);
  k_embed<<<NEDGE, 256, 0, stream>>>(bemb0, atom_edge_attr, ehbuf, 3, 9);

  for (int i = 0; i < 3; ++i) {
    const int last = (i == 2);
    // ===== atom-graph GINE =====
    cvt(ehbuf, abuf, (long)NEDGE * DD, stream);
    cvtT(ga_lin_w + (size_t)i * DD * DD, wbuf, DD, DD, stream);
    gemm(abuf, wbuf, ga_lin_b + (size_t)i * DD, tA, NEDGE, DD, DD, 0, stream);
    hipMemcpyAsync(tB, xbuf, (size_t)NNODE * DD * 4, hipMemcpyDeviceToDevice, stream);
    k_scatter<<<NEDGE, 256, 0, stream>>>(xbuf, tA, atom_ei, NEDGE, tB);
    cvt(tB, abuf, (long)NNODE * DD, stream);
    cvtT(ga_w1 + (size_t)i * DD * DD2, wbuf, DD, DD2, stream);
    gemm(abuf, wbuf, ga_b1 + (size_t)i * DD2, tA, NNODE, DD2, DD, 1, stream);
    cvt(tA, abuf, (long)NNODE * DD2, stream);
    cvtT(ga_w2 + (size_t)i * DD2 * DD, wbuf, DD2, DD, stream);
    gemm(abuf, wbuf, ga_b2 + (size_t)i * DD, tB, NNODE, DD, DD2, 0, stream);
    k_layernorm<<<NNODE, 256, 0, stream>>>(tB, ln_a_w + (size_t)i * DD, ln_a_b + (size_t)i * DD);
    zero(gstat, 2 * DD, stream);
    k_colstats<<<NNODE / 256, 256, 0, stream>>>(tB, gstat, NNODE);
    k_gn_apply<<<NNODE, 256, 0, stream>>>(tB, gstat, gn_a_w + (size_t)i * DD,
                                          gn_a_b + (size_t)i * DD, gn_a_ms + (size_t)i * DD,
                                          xbuf, NNODE, last);

    // ===== bond-graph GINE =====
    k_embed<<<NEDGE, 256, 0, stream>>>(bembL + (size_t)i * 3 * 9 * DD, atom_edge_attr, ceabuf, 3, 9);
    k_ang1<<<NANG, 256, 0, stream>>>(bond_attr, ang_w1 + (size_t)i * DD, ang_b1 + (size_t)i * DD, tA);
    cvt(tA, abuf, (long)NANG * DD, stream);
    cvtT(ang_w2 + (size_t)i * DD * DD, wbuf, DD, DD, stream);
    gemm(abuf, wbuf, ang_b2 + (size_t)i * DD, tB, NANG, DD, DD, 0, stream);   // ang
    cvt(tB, abuf, (long)NANG * DD, stream);
    cvtT(gb_lin_w + (size_t)i * DD * DD, wbuf, DD, DD, stream);
    gemm(abuf, wbuf, gb_lin_b + (size_t)i * DD, tA, NANG, DD, DD, 0, stream); // lin(ang)
    hipMemcpyAsync(tB, ceabuf, (size_t)NEDGE * DD * 4, hipMemcpyDeviceToDevice, stream);
    k_scatter<<<NANG, 256, 0, stream>>>(ceabuf, tA, bond_ei, NANG, tB);
    cvt(tB, abuf, (long)NEDGE * DD, stream);
    cvtT(gb_w1 + (size_t)i * DD * DD2, wbuf, DD, DD2, stream);
    gemm(abuf, wbuf, gb_b1 + (size_t)i * DD2, tA, NEDGE, DD2, DD, 1, stream);
    cvt(tA, abuf, (long)NEDGE * DD2, stream);
    cvtT(gb_w2 + (size_t)i * DD2 * DD, wbuf, DD2, DD, stream);
    gemm(abuf, wbuf, gb_b2 + (size_t)i * DD, tB, NEDGE, DD, DD2, 0, stream);
    k_layernorm<<<NEDGE, 256, 0, stream>>>(tB, ln_b_w + (size_t)i * DD, ln_b_b + (size_t)i * DD);
    zero(gstat, 2 * DD, stream);
    k_colstats<<<NEDGE / 256, 256, 0, stream>>>(tB, gstat, NEDGE);
    k_gn_apply<<<NEDGE, 256, 0, stream>>>(tB, gstat, gn_b_w + (size_t)i * DD,
                                          gn_b_b + (size_t)i * DD, gn_b_ms + (size_t)i * DD,
                                          ehbuf, NEDGE, last);
  }

  // ---- mean pooling + MLP head ----
  zero(sums, (long)NGRAPH * DD, stream);
  zero(cnt, NGRAPH, stream);
  k_pool<<<NNODE, 256, 0, stream>>>(xbuf, batch, sums, cnt);
  k_pooldiv<<<NGRAPH, 256, 0, stream>>>(sums, cnt);
  cvt(sums, abuf, (long)NGRAPH * DD, stream);
  cvtT(fc1_w, wbuf, DD, NFC, stream);
  gemm(abuf, wbuf, fc1_b, tA, NGRAPH, NFC, DD, 1, stream);
  cvt(tA, abuf, (long)NGRAPH * NFC, stream);
  cvtT(fc2_w, wbuf, NFC, DD, stream);
  gemm(abuf, wbuf, fc2_b, (float*)d_out, NGRAPH, DD, NFC, 0, stream);
}
